// TransitionModel_86062554677553
// MI455X (gfx1250) — compile-verified, matrix operand order checked by reading, not run on hardware
//
#include <hip/hip_runtime.h>

// ---------------- problem sizes ----------------
#define T_STEPS 64
#define BATCH   512
#define STATE   256
#define ACTION  64
#define BELIEF  1024
#define HIDDEN  1024
#define EMB     1024
#define MIN_STD 0.1f
#define SA_IN   (STATE + ACTION)       // 320
#define POS_IN  (BELIEF + EMB)         // 2048

typedef __attribute__((ext_vector_type(16))) __bf16 v16bf;
typedef __attribute__((ext_vector_type(8)))  __bf16 v8bf;
typedef __attribute__((ext_vector_type(8)))  float  v8f;

// ---------------- WMMA helpers ----------------
__device__ __forceinline__ v8f wmma_bf16(v16bf a, v16bf b, v8f c) {
    // D = A(16x32 bf16) x B(32x16 bf16) + C(16x16 f32)
    return __builtin_amdgcn_wmma_f32_16x16x32_bf16(
        /*neg_a=*/false, a, /*neg_b=*/false, b,
        /*c_mod=*/(short)0, c, /*reuse_a=*/false, /*reuse_b=*/false);
}

// Load one 16x32 bf16 fragment (A-operand layout; B-operand uses the same
// per-lane pattern with N playing the role of M, since B[k][n] = W[n][k] and
// W rows are contiguous along K).
// Lane L: row = row0 + (L&15), half = L>>4.
//   elements 0..7  <- K = k0 + half*8 + j        (VGPRs 0-3)
//   elements 8..15 <- K = k0 + 16 + half*8 + j   (VGPRs 4-7)
__device__ __forceinline__ v16bf load_frag(const __bf16* base, int ld,
                                           int row0, int k0, int lane) {
    int r = row0 + (lane & 15);
    int h = lane >> 4;
    const __bf16* p = base + (size_t)r * (size_t)ld + (size_t)(k0 + h * 8);
    v8bf lo = *reinterpret_cast<const v8bf*>(p);
    v8bf hi = *reinterpret_cast<const v8bf*>(p + 16);
    v16bf out;
#pragma unroll
    for (int i = 0; i < 8; ++i) { out[i] = lo[i]; out[i + 8] = hi[i]; }
    return out;
}

__device__ __forceinline__ float sigmoidf_(float x) {
    return 1.0f / (1.0f + expf(-x));
}
__device__ __forceinline__ float softplusf_(float x) {
    return (x > 20.0f) ? x : log1pf(expf(x));
}

// ---------------- kernels ----------------

// f32 -> bf16 copy
__global__ void cvt_f32_bf16(const float* __restrict__ src,
                             __bf16* __restrict__ dst, int n) {
    int i = blockIdx.x * blockDim.x + threadIdx.x;
    if (i < n) dst[i] = (__bf16)src[i];
}

// Per-step prologue: build X_sa = [s*m | a_t] (bf16) and write bf16(o_t)
// into right half of the posterior concat buffer.
__global__ void prestep_kernel(const __bf16* __restrict__ s_buf,
                               const float* __restrict__ A_t,
                               const float* __restrict__ M_t,
                               const float* __restrict__ O_t,
                               __bf16* __restrict__ Xsa,
                               __bf16* __restrict__ poscat) {
    int idx = blockIdx.x * blockDim.x + threadIdx.x;
    const int nx = BATCH * SA_IN;
    if (idx < nx) {
        int row = idx / SA_IN, col = idx - row * SA_IN;
        float v;
        if (col < STATE) {
            v = (float)s_buf[row * STATE + col] * M_t[row];
        } else {
            v = A_t[row * ACTION + (col - STATE)];
        }
        Xsa[idx] = (__bf16)v;
    } else {
        int j = idx - nx;
        if (j < BATCH * EMB) {
            int row = j / EMB, col = j - row * EMB;
            poscat[(size_t)row * POS_IN + BELIEF + col] = (__bf16)O_t[j];
        }
    }
}

// Generic GEMM: Out[512 x N] = relu(A[512 x K] @ W[N x K]^T + bias), bf16 out.
// block = 256 threads = 8 waves; wave w -> 32x32 register tile:
// rows (blockIdx.y*8+w)*32..+31, cols blockIdx.x*32..+31 (2x2 WMMA tiles;
// A- and B-fragments each reused twice -> 4 wmma per 4 fragment loads).
__global__ void __launch_bounds__(256)
gemm_bias_relu(const __bf16* __restrict__ A, const __bf16* __restrict__ W,
               const float* __restrict__ bias, __bf16* __restrict__ Out,
               int K, int N) {
    int lane = threadIdx.x & 31;
    int wav  = threadIdx.x >> 5;
    int m0 = (blockIdx.y * 8 + wav) * 32;
    int n0 = blockIdx.x * 32;
    v8f acc00 = {}, acc01 = {}, acc10 = {}, acc11 = {};
    for (int k0 = 0; k0 < K; k0 += 32) {
        v16bf a0 = load_frag(A, K, m0,      k0, lane);
        v16bf a1 = load_frag(A, K, m0 + 16, k0, lane);
        v16bf b0 = load_frag(W, K, n0,      k0, lane);
        v16bf b1 = load_frag(W, K, n0 + 16, k0, lane);
        acc00 = wmma_bf16(a0, b0, acc00);
        acc01 = wmma_bf16(a0, b1, acc01);
        acc10 = wmma_bf16(a1, b0, acc10);
        acc11 = wmma_bf16(a1, b1, acc11);
    }
    int c  = lane & 15;
    int rb = ((lane >> 4) << 3);
    float bi0 = bias[n0 + c];
    float bi1 = bias[n0 + 16 + c];
#pragma unroll
    for (int r = 0; r < 8; ++r) {
        int row0 = m0 + rb + r;
        int row1 = m0 + 16 + rb + r;
        Out[(size_t)row0 * N + n0 + c]      = (__bf16)fmaxf(acc00[r] + bi0, 0.0f);
        Out[(size_t)row0 * N + n0 + 16 + c] = (__bf16)fmaxf(acc01[r] + bi1, 0.0f);
        Out[(size_t)row1 * N + n0 + c]      = (__bf16)fmaxf(acc10[r] + bi0, 0.0f);
        Out[(size_t)row1 * N + n0 + 16 + c] = (__bf16)fmaxf(acc11[r] + bi1, 0.0f);
    }
}

// Fused GRU cell: 12 WMMA accumulators per wave (two M-tiles x {r,z,n} gates
// for x@Wih and h@Whh; weight fragments amortized across both M-tiles), then
// gate math. Writes f32 b_new to d_out, bf16 copies to the next-belief buffer
// and the posterior concat left half.
__global__ void __launch_bounds__(256)
gru_kernel(const __bf16* __restrict__ xb, const __bf16* __restrict__ hb,
           const __bf16* __restrict__ Wih, const __bf16* __restrict__ Whh,
           const float* __restrict__ bih, const float* __restrict__ bhh,
           const float* __restrict__ h_prev_f32,
           float* __restrict__ bnew_f32, __bf16* __restrict__ bnext,
           __bf16* __restrict__ poscat) {
    const int K = BELIEF;
    int lane = threadIdx.x & 31;
    int wav  = threadIdx.x >> 5;
    int m0 = (blockIdx.y * 8 + wav) * 32;
    int n0 = blockIdx.x * 16;
    v8f air0 = {}, aiz0 = {}, ain0 = {}, ahr0 = {}, ahz0 = {}, ahn0 = {};
    v8f air1 = {}, aiz1 = {}, ain1 = {}, ahr1 = {}, ahz1 = {}, ahn1 = {};
    for (int k0 = 0; k0 < K; k0 += 32) {
        v16bf ax0 = load_frag(xb, K, m0,      k0, lane);
        v16bf ax1 = load_frag(xb, K, m0 + 16, k0, lane);
        v16bf ah0 = load_frag(hb, K, m0,      k0, lane);
        v16bf ah1 = load_frag(hb, K, m0 + 16, k0, lane);
        v16bf wr = load_frag(Wih, K, n0,              k0, lane);
        v16bf wz = load_frag(Wih, K, n0 + BELIEF,     k0, lane);
        v16bf wn = load_frag(Wih, K, n0 + 2 * BELIEF, k0, lane);
        v16bf ur = load_frag(Whh, K, n0,              k0, lane);
        v16bf uz = load_frag(Whh, K, n0 + BELIEF,     k0, lane);
        v16bf un = load_frag(Whh, K, n0 + 2 * BELIEF, k0, lane);
        air0 = wmma_bf16(ax0, wr, air0);
        air1 = wmma_bf16(ax1, wr, air1);
        aiz0 = wmma_bf16(ax0, wz, aiz0);
        aiz1 = wmma_bf16(ax1, wz, aiz1);
        ain0 = wmma_bf16(ax0, wn, ain0);
        ain1 = wmma_bf16(ax1, wn, ain1);
        ahr0 = wmma_bf16(ah0, ur, ahr0);
        ahr1 = wmma_bf16(ah1, ur, ahr1);
        ahz0 = wmma_bf16(ah0, uz, ahz0);
        ahz1 = wmma_bf16(ah1, uz, ahz1);
        ahn0 = wmma_bf16(ah0, un, ahn0);
        ahn1 = wmma_bf16(ah1, un, ahn1);
    }
    int c  = n0 + (lane & 15);
    int rb = ((lane >> 4) << 3);
    float bir = bih[c], biz = bih[c + BELIEF], bin = bih[c + 2 * BELIEF];
    float bhr = bhh[c], bhz = bhh[c + BELIEF], bhn = bhh[c + 2 * BELIEF];
#pragma unroll
    for (int mt = 0; mt < 2; ++mt) {
        const v8f& air = mt ? air1 : air0;
        const v8f& aiz = mt ? aiz1 : aiz0;
        const v8f& ain = mt ? ain1 : ain0;
        const v8f& ahr = mt ? ahr1 : ahr0;
        const v8f& ahz = mt ? ahz1 : ahz0;
        const v8f& ahn = mt ? ahn1 : ahn0;
#pragma unroll
        for (int r = 0; r < 8; ++r) {
            int row = m0 + mt * 16 + rb + r;
            float i_r = air[r] + bir, h_r = ahr[r] + bhr;
            float i_z = aiz[r] + biz, h_z = ahz[r] + bhz;
            float i_n = ain[r] + bin, h_n = ahn[r] + bhn;
            float rg = sigmoidf_(i_r + h_r);
            float zg = sigmoidf_(i_z + h_z);
            float ng = tanhf(i_n + rg * h_n);
            float hp = h_prev_f32[(size_t)row * BELIEF + c];
            float bn = (1.0f - zg) * ng + zg * hp;
            bnew_f32[(size_t)row * BELIEF + c] = bn;
            bnext[(size_t)row * BELIEF + c]    = (__bf16)bn;
            poscat[(size_t)row * POS_IN + c]   = (__bf16)bn;
        }
    }
}

// Distribution heads: each wave computes two M-tiles of the paired
// (mu, pre-std) columns of the [512 x 512] head GEMM, then fuses
// softplus + reparameterized sampling. blockIdx.z: prior(0)/posterior(1).
__global__ void __launch_bounds__(256)
head_kernel(const __bf16* __restrict__ ph, const __bf16* __restrict__ qh,
            const __bf16* __restrict__ Wpri, const __bf16* __restrict__ Wpos,
            const float* __restrict__ bpri, const float* __restrict__ bpos,
            const float* __restrict__ eps_pri, const float* __restrict__ eps_pos,
            float* __restrict__ S_pri, float* __restrict__ MU_pri,
            float* __restrict__ STD_pri,
            float* __restrict__ S_pos, float* __restrict__ MU_pos,
            float* __restrict__ STD_pos, __bf16* __restrict__ s_buf) {
    const int K = HIDDEN;
    int z = blockIdx.z;
    const __bf16* A    = z ? qh   : ph;
    const __bf16* W    = z ? Wpos : Wpri;
    const float*  bias = z ? bpos : bpri;
    const float*  eps  = z ? eps_pos : eps_pri;
    float* S  = z ? S_pos  : S_pri;
    float* MU = z ? MU_pos : MU_pri;
    float* SD = z ? STD_pos : STD_pri;

    int lane = threadIdx.x & 31;
    int wav  = threadIdx.x >> 5;
    int m0 = (blockIdx.y * 8 + wav) * 32;
    int n0 = blockIdx.x * 16;                 // column within STATE (0..255)
    v8f amu0 = {}, ast0 = {}, amu1 = {}, ast1 = {};
    for (int k0 = 0; k0 < K; k0 += 32) {
        v16bf a0 = load_frag(A, K, m0,      k0, lane);
        v16bf a1 = load_frag(A, K, m0 + 16, k0, lane);
        v16bf wm = load_frag(W, K, n0,         k0, lane);
        v16bf ws = load_frag(W, K, n0 + STATE, k0, lane);
        amu0 = wmma_bf16(a0, wm, amu0);
        amu1 = wmma_bf16(a1, wm, amu1);
        ast0 = wmma_bf16(a0, ws, ast0);
        ast1 = wmma_bf16(a1, ws, ast1);
    }
    int c  = n0 + (lane & 15);
    int rb = ((lane >> 4) << 3);
    float bm = bias[c];
    float bs = bias[c + STATE];
#pragma unroll
    for (int mt = 0; mt < 2; ++mt) {
        const v8f& amu = mt ? amu1 : amu0;
        const v8f& ast = mt ? ast1 : ast0;
#pragma unroll
        for (int r = 0; r < 8; ++r) {
            int row = m0 + mt * 16 + rb + r;
            float mu = amu[r] + bm;
            float sd = softplusf_(ast[r] + bs) + MIN_STD;
            size_t o = (size_t)row * STATE + c;
            float sv = mu + sd * eps[o];
            MU[o] = mu;
            SD[o] = sd;
            S[o]  = sv;
            if (z) s_buf[o] = (__bf16)sv;
        }
    }
}

// ---------------- host launcher ----------------
extern "C" void kernel_launch(void* const* d_in, const int* in_sizes, int n_in,
                              void* d_out, int out_size, void* d_ws, size_t ws_size,
                              hipStream_t stream) {
    (void)in_sizes; (void)n_in; (void)out_size; (void)ws_size;
    // inputs (setup_inputs order)
    const float* s_0   = (const float*)d_in[0];
    const float* A_in  = (const float*)d_in[1];
    const float* b_0   = (const float*)d_in[2];
    const float* O_in  = (const float*)d_in[3];
    const float* M_in  = (const float*)d_in[4];
    const float* e_pri = (const float*)d_in[5];
    const float* e_pos = (const float*)d_in[6];
    const float* W_sa  = (const float*)d_in[7];
    const float* b_sa  = (const float*)d_in[8];
    const float* W_ih  = (const float*)d_in[9];
    const float* b_ih  = (const float*)d_in[10];
    const float* W_hh  = (const float*)d_in[11];
    const float* b_hh  = (const float*)d_in[12];
    const float* W_bpri = (const float*)d_in[13];
    const float* b_bpri = (const float*)d_in[14];
    const float* W_spri = (const float*)d_in[15];
    const float* b_spri = (const float*)d_in[16];
    const float* W_bpos = (const float*)d_in[17];
    const float* b_bpos = (const float*)d_in[18];
    const float* W_spos = (const float*)d_in[19];
    const float* b_spos = (const float*)d_in[20];

    float* out = (float*)d_out;
    const size_t stepB = (size_t)BATCH * BELIEF;           // 524288
    const size_t stepS = (size_t)BATCH * STATE;            // 131072
    float* O_B      = out;
    float* O_Spri   = out + (size_t)T_STEPS * stepB;
    float* O_MUpri  = O_Spri  + (size_t)T_STEPS * stepS;
    float* O_STDpri = O_MUpri + (size_t)T_STEPS * stepS;
    float* O_Spos   = O_STDpri + (size_t)T_STEPS * stepS;
    float* O_MUpos  = O_Spos  + (size_t)T_STEPS * stepS;
    float* O_STDpos = O_MUpos + (size_t)T_STEPS * stepS;

    // workspace layout (bf16 elements, 64-elem aligned)
    __bf16* base = (__bf16*)d_ws;
    size_t off = 0;
    auto alloc = [&](size_t n) { size_t o = off; off += (n + 63) & ~(size_t)63; return o; };
    __bf16* wsa   = base + alloc((size_t)BELIEF * SA_IN);
    __bf16* wih   = base + alloc((size_t)3 * BELIEF * BELIEF);
    __bf16* whh   = base + alloc((size_t)3 * BELIEF * BELIEF);
    __bf16* wbpri = base + alloc((size_t)HIDDEN * BELIEF);
    __bf16* wspri = base + alloc((size_t)2 * STATE * HIDDEN);
    __bf16* wbpos = base + alloc((size_t)HIDDEN * POS_IN);
    __bf16* wspos = base + alloc((size_t)2 * STATE * HIDDEN);
    __bf16* xsa   = base + alloc((size_t)BATCH * SA_IN);
    __bf16* xbuf  = base + alloc((size_t)BATCH * BELIEF);
    __bf16* bbuf0 = base + alloc((size_t)BATCH * BELIEF);
    __bf16* bbuf1 = base + alloc((size_t)BATCH * BELIEF);
    __bf16* sbuf  = base + alloc((size_t)BATCH * STATE);
    __bf16* phb   = base + alloc((size_t)BATCH * HIDDEN);
    __bf16* qhb   = base + alloc((size_t)BATCH * HIDDEN);
    __bf16* pcat  = base + alloc((size_t)BATCH * POS_IN);

    // one-time conversions (re-run every call; deterministic)
    auto cvt = [&](const float* s, __bf16* d, size_t n) {
        cvt_f32_bf16<<<dim3((unsigned)((n + 255) / 256)), dim3(256), 0, stream>>>(s, d, (int)n);
    };
    cvt(W_sa,   wsa,   (size_t)BELIEF * SA_IN);
    cvt(W_ih,   wih,   (size_t)3 * BELIEF * BELIEF);
    cvt(W_hh,   whh,   (size_t)3 * BELIEF * BELIEF);
    cvt(W_bpri, wbpri, (size_t)HIDDEN * BELIEF);
    cvt(W_spri, wspri, (size_t)2 * STATE * HIDDEN);
    cvt(W_bpos, wbpos, (size_t)HIDDEN * POS_IN);
    cvt(W_spos, wspos, (size_t)2 * STATE * HIDDEN);
    cvt(b_0,    bbuf0, (size_t)BATCH * BELIEF);
    cvt(s_0,    sbuf,  (size_t)BATCH * STATE);

    const dim3 blk(256);
    const dim3 gGemm(BELIEF / 32, BATCH / 256);     // N=1024 gemms: 32x2 blocks
    const dim3 gGru(BELIEF / 16, BATCH / 256);      // 64x2 blocks
    const dim3 gHead(STATE / 16, BATCH / 256, 2);   // 16x2x2 blocks
    const unsigned nPre = (unsigned)((BATCH * SA_IN + BATCH * EMB + 255) / 256);

    __bf16* bcur = bbuf0;
    __bf16* bnxt = bbuf1;
    for (int t = 0; t < T_STEPS; ++t) {
        const float* A_t = A_in + (size_t)t * BATCH * ACTION;
        const float* O_t = O_in + (size_t)t * BATCH * EMB;
        const float* M_t = M_in + (size_t)t * BATCH;
        const float* ep  = e_pri + (size_t)t * stepS;
        const float* eq  = e_pos + (size_t)t * stepS;
        const float* h_prev = (t == 0) ? b_0 : (O_B + (size_t)(t - 1) * stepB);

        prestep_kernel<<<dim3(nPre), blk, 0, stream>>>(sbuf, A_t, M_t, O_t, xsa, pcat);

        gemm_bias_relu<<<gGemm, blk, 0, stream>>>(xsa, wsa, b_sa, xbuf,
                                                  SA_IN, BELIEF);

        gru_kernel<<<gGru, blk, 0, stream>>>(xbuf, bcur, wih, whh, b_ih, b_hh,
                                             h_prev, O_B + (size_t)t * stepB,
                                             bnxt, pcat);

        gemm_bias_relu<<<gGemm, blk, 0, stream>>>(bnxt, wbpri, b_bpri, phb,
                                                  BELIEF, HIDDEN);
        gemm_bias_relu<<<gGemm, blk, 0, stream>>>(pcat, wbpos, b_bpos, qhb,
                                                  POS_IN, HIDDEN);

        head_kernel<<<gHead, blk, 0, stream>>>(
            phb, qhb, wspri, wspos, b_spri, b_spos, ep, eq,
            O_Spri  + (size_t)t * stepS, O_MUpri + (size_t)t * stepS,
            O_STDpri + (size_t)t * stepS,
            O_Spos  + (size_t)t * stepS, O_MUpos + (size_t)t * stepS,
            O_STDpos + (size_t)t * stepS, sbuf);

        __bf16* tmp = bcur; bcur = bnxt; bnxt = tmp;
    }
}
